// multi_head_attn_28200755265969
// MI455X (gfx1250) — matrix operand order repaired and checked by reading, not compiled, not run on hardware
//
#include <hip/hip_runtime.h>

typedef _Float16 v16h __attribute__((ext_vector_type(16)));
typedef _Float16 h8   __attribute__((ext_vector_type(8)));
typedef float    v8f  __attribute__((ext_vector_type(8)));

#define S_SEQ   2048
#define E_EMB   1024
#define H_HEADS 16
#define DH      64
#define WIN     256

__device__ __forceinline__ v8f wmma_f16(v16h a, v16h b, v8f c) {
    // 8 args: (neg_a, A, neg_b, B, c_mod, C, reuse_a, reuse_b)
    return __builtin_amdgcn_wmma_f32_16x16x32_f16(false, a, false, b, (short)0, c,
                                                  false, false);
}

// A-fragment loader: 16x32 A in contiguous k order (same as the B loads) — lane holds row (lane&15),
// 16 contiguous K starting at (sel*16).
__device__ __forceinline__ v16h load_afrag(const _Float16* row, int sel) {
    union { v16h v; h8 p[2]; } a;
    a.p[0] = *(const h8*)(row + sel * 16);
    a.p[1] = *(const h8*)(row + sel * 16 + 8);
    return a.v;
}

// ---------------------------------------------------------------------------
// Stage 0: fp32 -> f16 conversion (8 elems/thread)
// ---------------------------------------------------------------------------
__global__ void cvt_f32_f16_kernel(const float* __restrict__ src,
                                   _Float16* __restrict__ dst, int n) {
    int i = (blockIdx.x * blockDim.x + threadIdx.x) * 8;
    if (i + 7 < n) {
        float4 f0 = *(const float4*)(src + i);
        float4 f1 = *(const float4*)(src + i + 4);
        h8 o;
        o[0] = (_Float16)f0.x; o[1] = (_Float16)f0.y;
        o[2] = (_Float16)f0.z; o[3] = (_Float16)f0.w;
        o[4] = (_Float16)f1.x; o[5] = (_Float16)f1.y;
        o[6] = (_Float16)f1.z; o[7] = (_Float16)f1.w;
        *(h8*)(dst + i) = o;
    }
}

// ---------------------------------------------------------------------------
// Stage 1: Q/K/V projections. One wave per 32(s) x 64(dh) tile of one matrix.
//   2 A-fragments x 4 B-fragments -> 8 WMMAs per K=32 step (A/B reuse).
//   Q,K stored [H,S,DH]; V stored transposed [H,DH,S].
// ---------------------------------------------------------------------------
__global__ __launch_bounds__(256)
void proj_kernel(const _Float16* __restrict__ X,    // [S,E]
                 const _Float16* __restrict__ Wq,   // [H,DH,E]
                 const _Float16* __restrict__ Wk,
                 const _Float16* __restrict__ Wv,
                 _Float16* __restrict__ Q,          // [H,S,DH]
                 _Float16* __restrict__ Kd,         // [H,S,DH]
                 _Float16* __restrict__ Vt)         // [H,DH,S]
{
    int wave = blockIdx.x * (blockDim.x >> 5) + (threadIdx.x >> 5);
    int lane = threadIdx.x & 31;
    int col  = lane & 15;
    int sel  = lane >> 4;

    // 3 matrices * 16 heads * 64 s-blocks(32 rows) = 3072 waves
    const int SPB = S_SEQ / 32;                 // 64
    int m  = wave / (H_HEADS * SPB);            // 0=Q 1=K 2=V
    int r  = wave % (H_HEADS * SPB);
    int h  = r / SPB;
    int s0 = (r % SPB) * 32;

    const _Float16* W = (m == 0) ? Wq : (m == 1) ? Wk : Wv;

    const _Float16* xrow0 = X + (size_t)(s0 + col) * E_EMB;
    const _Float16* xrow1 = X + (size_t)(s0 + 16 + col) * E_EMB;
    const _Float16* wbase = W + (size_t)h * DH * E_EMB;

    v8f acc[2][4] = {};
    for (int kk = 0; kk < E_EMB; kk += 32) {
        v16h a0 = load_afrag(xrow0 + kk, sel);
        v16h a1 = load_afrag(xrow1 + kk, sel);
        #pragma unroll
        for (int t = 0; t < 4; ++t) {
            // B column n = dh index t*16+col, 16 contiguous K per lane
            v16h b = *(const v16h*)(wbase + (size_t)(t * 16 + col) * E_EMB +
                                    kk + sel * 16);
            acc[0][t] = wmma_f16(a0, b, acc[0][t]);
            acc[1][t] = wmma_f16(a1, b, acc[1][t]);
        }
    }

    if (m == 2) {
        // Vt[h][t*16+col][s0 + mi*16 + sel*8 + rr] : contiguous h8 per lane
        #pragma unroll
        for (int mi = 0; mi < 2; ++mi)
            #pragma unroll
            for (int t = 0; t < 4; ++t) {
                h8 o;
                #pragma unroll
                for (int rr = 0; rr < 8; ++rr) o[rr] = (_Float16)acc[mi][t][rr];
                *(h8*)(Vt + ((size_t)h * DH + t * 16 + col) * S_SEQ +
                       s0 + mi * 16 + sel * 8) = o;
            }
    } else {
        _Float16* dst = (m == 0) ? Q : Kd;
        #pragma unroll
        for (int mi = 0; mi < 2; ++mi)
            #pragma unroll
            for (int t = 0; t < 4; ++t)
                #pragma unroll
                for (int rr = 0; rr < 8; ++rr) {
                    int srow = s0 + mi * 16 + rr + sel * 8;
                    dst[((size_t)h * S_SEQ + srow) * DH + t * 16 + col] =
                        (_Float16)acc[mi][t][rr];
                }
    }
}

// ---------------------------------------------------------------------------
// Stage 2: sliding-window flash attention. One wave per (head, 16-query tile).
// ---------------------------------------------------------------------------
__global__ __launch_bounds__(32)
void attn_kernel(const _Float16* __restrict__ Q,    // [H,S,DH]
                 const _Float16* __restrict__ Kd,   // [H,S,DH]
                 const _Float16* __restrict__ Vt,   // [H,DH,S]
                 _Float16* __restrict__ delta)      // [S, H*DH]
{
    __shared__ _Float16 lp[16][34];   // P tile, padded rows (17-dword stride)

    int q0   = blockIdx.x * 16;
    int h    = blockIdx.y;
    int lane = threadIdx.x & 31;
    int col  = lane & 15;
    int sel  = lane >> 4;

    // Q A-fragments for the whole tile (dh = 64 -> two K=32 chunks), kept live.
    const _Float16* qrow = Q + ((size_t)h * S_SEQ + q0 + col) * DH;
    v16h aq0 = load_afrag(qrow, sel);
    v16h aq1 = load_afrag(qrow + 32, sel);

    float mr[8], lr[8];
    v8f O[4] = {};
    #pragma unroll
    for (int rr = 0; rr < 8; ++rr) { mr[rr] = -3.0e38f; lr[rr] = 0.0f; }

    int klo = q0 - WIN;      if (klo < 0) klo = 0;
    int khi = q0 + 16 + WIN; if (khi > S_SEQ) khi = S_SEQ;
    const float scale = 0.125f;   // 1/sqrt(64)

    for (int kb = klo; kb < khi; kb += 32) {
        // ---- scores: two 16x16 tiles (keys kb..kb+31), K-loop over dh ----
        v8f sc[2];
        #pragma unroll
        for (int t = 0; t < 2; ++t) {
            int key  = kb + t * 16 + col;
            int keyc = key < S_SEQ ? key : S_SEQ - 1;       // clamp OOB loads
            const _Float16* kcol = Kd + ((size_t)h * S_SEQ + keyc) * DH;
            v16h b0 = *(const v16h*)(kcol + sel * 16);
            v16h b1 = *(const v16h*)(kcol + 32 + sel * 16);
            v8f c = {};
            c = wmma_f16(aq0, b0, c);
            c = wmma_f16(aq1, b1, c);
            sc[t] = c;
        }
        // ---- scale + sliding-window mask:  j-w <= k < j+w, k != j, k < S ----
        #pragma unroll
        for (int t = 0; t < 2; ++t) {
            int key = kb + t * 16 + col;
            #pragma unroll
            for (int rr = 0; rr < 8; ++rr) {
                int j = q0 + rr + sel * 8;
                bool valid = (key >= j - WIN) & (key < j + WIN) &
                             (key != j) & (key < S_SEQ);
                sc[t][rr] = valid ? sc[t][rr] * scale : -3.0e38f;
            }
        }
        // ---- online softmax: row max / rescale / row sum ----
        float alpha[8];
        #pragma unroll
        for (int rr = 0; rr < 8; ++rr) {
            float v = fmaxf(sc[0][rr], sc[1][rr]);
            v = fmaxf(v, __shfl_xor(v, 1, 32));
            v = fmaxf(v, __shfl_xor(v, 2, 32));
            v = fmaxf(v, __shfl_xor(v, 4, 32));
            v = fmaxf(v, __shfl_xor(v, 8, 32));   // stays inside 16-lane half
            float mn = fmaxf(mr[rr], v);
            alpha[rr] = __expf(mr[rr] - mn);
            mr[rr] = mn;
        }
        #pragma unroll
        for (int rr = 0; rr < 8; ++rr) {
            float p0 = __expf(sc[0][rr] - mr[rr]);
            float p1 = __expf(sc[1][rr] - mr[rr]);
            sc[0][rr] = p0; sc[1][rr] = p1;
            float v = p0 + p1;
            v += __shfl_xor(v, 1, 32);
            v += __shfl_xor(v, 2, 32);
            v += __shfl_xor(v, 4, 32);
            v += __shfl_xor(v, 8, 32);
            lr[rr] = lr[rr] * alpha[rr] + v;
        }
        #pragma unroll
        for (int t = 0; t < 4; ++t)
            #pragma unroll
            for (int rr = 0; rr < 8; ++rr) O[t][rr] *= alpha[rr];

        // ---- P: C-layout -> LDS (row-major 16x32) -> A-layout fragment ----
        __syncthreads();
        #pragma unroll
        for (int rr = 0; rr < 8; ++rr) {
            int row = rr + sel * 8;
            lp[row][col]      = (_Float16)sc[0][rr];
            lp[row][col + 16] = (_Float16)sc[1][rr];
        }
        __syncthreads();
        union { v16h v; unsigned int u[8]; } ap;
        {
            int row = col;          // A row = lane % 16
            int b0  = sel * 16;
            #pragma unroll
            for (int i2 = 0; i2 < 4; ++i2) {
                ap.u[i2]     = *(const unsigned int*)&lp[row][b0 + 2 * i2];
                ap.u[4 + i2] = *(const unsigned int*)&lp[row][b0 + 8 + 2 * i2];
            }
        }
        // ---- O += P @ V  (B column n = dh-dim, contiguous over keys) ----
        int kload = kb + sel * 16;
        if (kload > S_SEQ - 16) kload = S_SEQ - 16;   // clamp; masked P is 0
        #pragma unroll
        for (int t = 0; t < 4; ++t) {
            const _Float16* vptr =
                Vt + ((size_t)h * DH + t * 16 + col) * S_SEQ + kload;
            v16h bv = *(const v16h*)vptr;
            O[t] = wmma_f16(ap.v, bv, O[t]);
        }
    }

    // ---- normalize and store delta as [S, H*DH] ----
    #pragma unroll
    for (int t = 0; t < 4; ++t) {
        #pragma unroll
        for (int rr = 0; rr < 8; ++rr) {
            float o  = O[t][rr] / lr[rr];
            int srow = q0 + rr + sel * 8;
            delta[(size_t)srow * (H_HEADS * DH) + h * DH + t * 16 + col] =
                (_Float16)o;
        }
    }
}

// ---------------------------------------------------------------------------
// Stage 3: out = input + delta @ Wup^T   (M=S, N=E, K=H*DH=1024)
//   One wave per 32(s) x 64(e) tile: 2 A x 4 B fragments -> 8 WMMAs / K-step.
// ---------------------------------------------------------------------------
__global__ __launch_bounds__(256)
void out_kernel(const _Float16* __restrict__ delta,  // [S, H*DH]
                const _Float16* __restrict__ Wup,    // [H,E,DH] f16
                const float* __restrict__ X,         // [S,E] original fp32
                float* __restrict__ out)             // [S,E]
{
    int wave = blockIdx.x * (blockDim.x >> 5) + (threadIdx.x >> 5);
    int lane = threadIdx.x & 31;
    int col  = lane & 15;
    int sel  = lane >> 4;
    const int EBLK = E_EMB / 64;                // 16 e-blocks of 64
    int s0 = (wave / EBLK) * 32;
    int e0 = (wave % EBLK) * 64;

    const _Float16* arow0 = delta + (size_t)(s0 + col) * (H_HEADS * DH);
    const _Float16* arow1 = delta + (size_t)(s0 + 16 + col) * (H_HEADS * DH);

    v8f acc[2][4] = {};
    for (int kk = 0; kk < H_HEADS * DH; kk += 32) {
        v16h a0 = load_afrag(arow0 + kk, sel);
        v16h a1 = load_afrag(arow1 + kk, sel);
        int kbB = kk + sel * 16;
        int hh  = kbB >> 6;
        int dd  = kbB & 63;
        #pragma unroll
        for (int t = 0; t < 4; ++t) {
            // B[k][e] with k = h*64 + d : 16 contiguous d within one head
            int e = e0 + t * 16 + col;
            v16h b = *(const v16h*)(Wup + ((size_t)hh * E_EMB + e) * DH + dd);
            acc[0][t] = wmma_f16(a0, b, acc[0][t]);
            acc[1][t] = wmma_f16(a1, b, acc[1][t]);
        }
    }
    #pragma unroll
    for (int mi = 0; mi < 2; ++mi)
        #pragma unroll
        for (int t = 0; t < 4; ++t)
            #pragma unroll
            for (int rr = 0; rr < 8; ++rr) {
                int srow   = s0 + mi * 16 + rr + sel * 8;
                size_t idx = (size_t)srow * E_EMB + e0 + t * 16 + col;
                out[idx] = X[idx] + acc[mi][t][rr];
            }
}

// ---------------------------------------------------------------------------
// Launcher
// ---------------------------------------------------------------------------
extern "C" void kernel_launch(void* const* d_in, const int* in_sizes, int n_in,
                              void* d_out, int out_size, void* d_ws, size_t ws_size,
                              hipStream_t stream) {
    const float* x   = (const float*)d_in[0];   // [S,E]
    const float* wq  = (const float*)d_in[1];   // [H,DH,E]
    const float* wk  = (const float*)d_in[2];   // [H,DH,E]
    const float* wv  = (const float*)d_in[3];   // [H,DH,E]
    const float* wup = (const float*)d_in[4];   // [H,E,DH]
    float* out = (float*)d_out;

    char* ws = (char*)d_ws;
    const size_t MiB = 1024u * 1024u;
    _Float16* X16  = (_Float16*)(ws + 0 * MiB);    // 4 MiB  [S,E]
    _Float16* Wq16 = (_Float16*)(ws + 4 * MiB);    // 2 MiB
    _Float16* Wk16 = (_Float16*)(ws + 6 * MiB);    // 2 MiB
    _Float16* Wv16 = (_Float16*)(ws + 8 * MiB);    // 2 MiB
    _Float16* Wu16 = (_Float16*)(ws + 10 * MiB);   // 2 MiB
    _Float16* Q16  = (_Float16*)(ws + 12 * MiB);   // 4 MiB  [H,S,DH]
    _Float16* K16  = (_Float16*)(ws + 16 * MiB);   // 4 MiB  [H,S,DH]
    _Float16* Vt16 = (_Float16*)(ws + 20 * MiB);   // 4 MiB  [H,DH,S]
    _Float16* D16  = (_Float16*)(ws + 24 * MiB);   // 4 MiB  [S,H*DH]

    const int nX = S_SEQ * E_EMB;           // 2,097,152
    const int nW = H_HEADS * DH * E_EMB;    // 1,048,576

    cvt_f32_f16_kernel<<<nX / (256 * 8), 256, 0, stream>>>(x,   X16, nX);
    cvt_f32_f16_kernel<<<nW / (256 * 8), 256, 0, stream>>>(wq,  Wq16, nW);
    cvt_f32_f16_kernel<<<nW / (256 * 8), 256, 0, stream>>>(wk,  Wk16, nW);
    cvt_f32_f16_kernel<<<nW / (256 * 8), 256, 0, stream>>>(wv,  Wv16, nW);
    cvt_f32_f16_kernel<<<nW / (256 * 8), 256, 0, stream>>>(wup, Wu16, nW);

    // 3 matrices * 16 heads * 64 s-blocks = 3072 waves / 8 per block
    proj_kernel<<<384, 256, 0, stream>>>(X16, Wq16, Wk16, Wv16, Q16, K16, Vt16);

    attn_kernel<<<dim3(S_SEQ / 16, H_HEADS), 32, 0, stream>>>(Q16, K16, Vt16, D16);

    // 64 s-blocks * 16 e-blocks = 1024 waves / 8 per block
    out_kernel<<<128, 256, 0, stream>>>(D16, Wu16, x, out);
}
